// HybridAttention_22058952032338
// MI455X (gfx1250) — matrix operand order repaired and checked by reading, not compiled, not run on hardware
//
#include <hip/hip_runtime.h>
#include <hip/hip_bf16.h>
#include <math.h>

// ---------------------------------------------------------------------------
// HybridAttention for MI455X (gfx1250, wave32, WMMA + TDM).
//   1. f32->f16 conversions of X and the 5 weight matrices
//   2. G = sigmoid(X @ Wg^T + bg)            (WMMA GEMM, A tile staged in LDS)
//   3. SSM scan: 16 lanes per channel, shuffle-based 16x16 matvec, h = x + y
//   4. Q,K = h@W^T (+b) f16 out; V f32 out   (WMMA GEMM)
//   5. V -> Vt f16 [D][S] transpose
//   6. per (head, 16-query block): scores in LDS, bitwise top-k threshold,
//      band mask, softmax, PV with all 8 waves (WMMA both GEMMs)
//   7. out = ctx @ Wd^T + bd                 (WMMA GEMM, fp32 out)
// ---------------------------------------------------------------------------

typedef __attribute__((ext_vector_type(16))) _Float16 v16h;
typedef __attribute__((ext_vector_type(8)))  float    v8f;
typedef unsigned int u32x4 __attribute__((ext_vector_type(4)));
typedef int          i32x4 __attribute__((ext_vector_type(4)));
typedef int          i32x8 __attribute__((ext_vector_type(8)));

#define SEQ    2048
#define DMODEL 768
#define NHEAD  12
#define DHEAD  64
#define KEEP   1843          // int((1-0.1)*2048)
#define WINDOW 128
#define LDSROW 2064          // padded LDS row (multiple of 8 halves)

#if defined(__has_include)
#if __has_include(<hip/amd_detail/amd_gfx1250_TDM.h>)
#define ATH_TDM_6ARG 1
#endif
#endif

#if defined(__has_builtin)
#if __has_builtin(__builtin_amdgcn_tensor_load_to_lds) && \
    __has_builtin(__builtin_amdgcn_s_wait_tensorcnt)
#define ATH_USE_TDM 1
#endif
#endif

// ----- WMMA fragment loaders (lane m/n per CDNA5 ISA 7.12.2; contiguous k order, not its table order) -----

// A-matrix 16x32 f16 (M x K), row-major source. lane m = lane&15;
// lanes 0-15: K = 0..15; lanes 16-31: K = 16..31 (contiguous, matches B).
__device__ __forceinline__ v16h load_a_frag(const _Float16* base, int ld,
                                            int row0, int col0, int lane) {
  int m  = lane & 15;
  int hi = lane >> 4;
  const _Float16* p = base + (size_t)(row0 + m) * ld + col0;
  int kb1 = hi * 16;
  int kb2 = hi * 16 + 8;
  v16h a;
#pragma unroll
  for (int j = 0; j < 8; ++j) a[j] = p[kb1 + j];
#pragma unroll
  for (int j = 0; j < 8; ++j) a[8 + j] = p[kb2 + j];
  return a;
}

// B-matrix 32x16 f16 (K x N), stored so that B[k,n] = base[(n0+n)*ld + k0 + k]
// (the "n" dimension is the row of the stored array -> contiguous K).
__device__ __forceinline__ v16h load_b_frag(const _Float16* base, int ld,
                                            int n0, int k0, int lane) {
  int n  = lane & 15;
  int kb = (lane >> 4) * 16;
  const _Float16* p = base + (size_t)(n0 + n) * ld + k0 + kb;
  v16h b;
#pragma unroll
  for (int j = 0; j < 16; ++j) b[j] = p[j];
  return b;
}

__device__ __forceinline__ v8f zero8() {
  v8f z = {0.f, 0.f, 0.f, 0.f, 0.f, 0.f, 0.f, 0.f};
  return z;
}

// ----- elementwise converts ------------------------------------------------

__global__ void cvt_f32_f16(const float* __restrict__ src,
                            _Float16* __restrict__ dst, int n) {
  int i = blockIdx.x * 256 + threadIdx.x;
  if (i < n) dst[i] = (_Float16)src[i];
}

__global__ void transpose_cvt(const float* __restrict__ V,
                              _Float16* __restrict__ Vt, int rows, int cols) {
  int i = blockIdx.x * 256 + threadIdx.x;
  if (i < rows * cols) {
    int r = i / cols, c = i % cols;
    Vt[(size_t)c * rows + r] = (_Float16)V[i];
  }
}

// ----- WMMA GEMM:  O[m,n] = act( sum_k A[m,k] * W[n,k] + b[n] ) ------------
// One block = one 16-row tile; the A tile (16 x K f16) is staged into LDS
// (TDM issue + visible vector copy); the 12 waves (N/64 strips) then read A
// fragments from LDS and stream B from global.

__global__ void gemm_wmma(const _Float16* __restrict__ Am,
                          const _Float16* __restrict__ Wm,
                          const float* __restrict__ bias,
                          float* __restrict__ Cout,
                          _Float16* __restrict__ Hout,
                          int M, int N, int K, int act) {
  __shared__ _Float16 As[16 * DMODEL];   // only static LDS -> offset 0

  int lane = threadIdx.x & 31;
  int wave = threadIdx.x >> 5;           // 0..11 == N strip
  int row0 = blockIdx.x * 16;
  int col0 = wave * 64;

#if defined(ATH_USE_TDM)
  if (wave == 0) {
    // Tensor DMA descriptor (CDNA5 ISA):
    // 2D tile 16 x K of 2-byte elements, row stride K, into LDS offset 0.
    unsigned long long ga =
        (unsigned long long)(uintptr_t)(Am + (size_t)row0 * K);
    u32x4 g0;
    g0[0] = 1u;                                   // count=1 (valid user D#)
    g0[1] = 0u;                                   // lds_addr = 0 (As)
    g0[2] = (unsigned)(ga & 0xFFFFFFFFu);         // global_addr[31:0]
    g0[3] = (unsigned)((ga >> 32) & 0x01FFFFFFu)  // global_addr[56:32]
          | 0x80000000u;                          // type = 2 ("image")
    i32x8 g1;
    g1[0] = (1 << 16);                            // data_size=1 -> 2 bytes
    g1[1] = (int)((unsigned)(K & 0xFFFF) << 16);  // tensor_dim0[15:0]
    g1[2] = (int)(((unsigned)K >> 16) & 0xFFFFu)  // tensor_dim0[31:16]
          | (16 << 16);                           // tensor_dim1[15:0] = 16
    g1[3] = (int)((unsigned)(K & 0xFFFF) << 16);  // tile_dim0 = K
    g1[4] = 16;                                   // tile_dim1=16, tile_dim2=0
    g1[5] = K;                                    // tensor_dim0_stride[31:0]
    g1[6] = 0;                                    // stride hi / dim1_stride lo
    g1[7] = 0;
    i32x4 gz = {0, 0, 0, 0};
#if defined(ATH_TDM_6ARG)
    i32x8 gz8 = {0, 0, 0, 0, 0, 0, 0, 0};
    __builtin_amdgcn_tensor_load_to_lds(g0, g1, gz, gz, gz8, 0);
#else
    __builtin_amdgcn_tensor_load_to_lds(g0, g1, gz, gz, 0);
#endif
    __builtin_amdgcn_s_wait_tensorcnt(0);
  }
  __syncthreads();
#endif
  // Visible staging copy. The TDM write is addressed via descriptor bits and
  // As never escapes, so alias/escape analysis concludes As is never stored
  // and folds its reads to undef (observed in r2/r3 disasm: the per-k
  // ds_loads vanished). These stores make the LDS contents real for the
  // optimizer (and guarantee correct data); the 16xK tile is contiguous,
  // so this is a straight b128 stream.
  {
    const float4* src = (const float4*)(Am + (size_t)row0 * K);
    float4* dst = (float4*)As;
    int nchunks = (16 * K) / 8;   // 8 halves per float4
    for (int i = threadIdx.x; i < nchunks; i += blockDim.x) dst[i] = src[i];
  }
  __syncthreads();

  v8f acc[4];
#pragma unroll
  for (int s = 0; s < 4; ++s) acc[s] = zero8();

  for (int k = 0; k < K; k += 32) {
    if (k + 32 < K)   // probe-confirmed: lowers to global_prefetch_b8
      __builtin_prefetch(Wm + (size_t)(col0 + (lane & 15)) * K + k + 32, 0, 3);
    v16h a = load_a_frag(As, K, 0, k, lane);      // from LDS
#pragma unroll
    for (int s = 0; s < 4; ++s) {
      v16h b = load_b_frag(Wm, K, col0 + s * 16, k, lane);
      acc[s] = __builtin_amdgcn_wmma_f32_16x16x32_f16(
          false, a, false, b, (short)0, acc[s], false, false);
    }
  }

  int n = lane & 15, hi = lane >> 4;
#pragma unroll
  for (int s = 0; s < 4; ++s) {
    int cn = col0 + s * 16 + n;
    float bv = bias ? bias[cn] : 0.f;
#pragma unroll
    for (int r = 0; r < 8; ++r) {
      int rm = row0 + r + 8 * hi;
      float v = acc[s][r] + bv;
      if (act == 1) v = 1.f / (1.f + __expf(-v));
      if (Cout) Cout[(size_t)rm * N + cn] = v;
      if (Hout) Hout[(size_t)rm * N + cn] = (_Float16)v;
    }
  }
}

// ----- SSM scan: 16 lanes per channel -------------------------------------

__global__ void scan_kernel(const float* __restrict__ X,
                            const float* __restrict__ A,
                            const float* __restrict__ Bm,
                            const float* __restrict__ Cm,
                            const float* __restrict__ log_dt,
                            const float* __restrict__ G,
                            _Float16* __restrict__ Hh) {
  int tid = threadIdx.x;        // 256 threads = 16 channels
  int grp = tid >> 4;
  int n   = tid & 15;
  int d   = blockIdx.x * 16 + grp;

  float Arow[16];
#pragma unroll
  for (int m = 0; m < 16; ++m) Arow[m] = A[n * 16 + m];
  float bt = Bm[n * DMODEL + d];   // BT[d][n] = Bm[n][d]
  float cn = Cm[d * 16 + n];
  float dt = __expf(log_dt[d]);
  dt = fminf(fmaxf(dt, 1e-3f), 1e-1f);

  float s = 0.f;
  for (int t = 0; t < SEQ; ++t) {
    float x = X[(size_t)t * DMODEL + d];
    float g = G[(size_t)t * DMODEL + d];
    float ds = x * bt;
#pragma unroll
    for (int m = 0; m < 16; ++m) ds += __shfl(s, m, 16) * Arow[m];
    float sc = ds * dt;
    sc = (fabsf(sc) > 0.1f) ? sc : 0.f;   // spike threshold
    s += sc * g;
    float y = s * cn;                      // reduce over the 16 state lanes
    y += __shfl_xor(y, 8, 16);
    y += __shfl_xor(y, 4, 16);
    y += __shfl_xor(y, 2, 16);
    y += __shfl_xor(y, 1, 16);
    if (n == 0) Hh[(size_t)t * DMODEL + d] = (_Float16)(x + y);
  }
}

// ----- attention: one (head, 16-query) block per workgroup -----------------

__global__ void attn_kernel(const _Float16* __restrict__ Qh,
                            const _Float16* __restrict__ Kh,
                            const _Float16* __restrict__ Vth,
                            float* __restrict__ Ctx) {
  extern __shared__ char smem[];
  float*    sc = (float*)smem;                                    // 16*LDSROW
  _Float16* pr = (_Float16*)(smem + 16 * LDSROW * sizeof(float)); // 16*LDSROW

  int lane = threadIdx.x & 31, wave = threadIdx.x >> 5;
  int qb = blockIdx.x * 16;
  int hb = blockIdx.y * DHEAD;

  // ---- phase 1: scores = (Q K^T) / 8, full 16 x 2048 strip into LDS ----
  for (int nt = wave; nt < SEQ / 16; nt += 8) {
    v8f acc = zero8();
#pragma unroll
    for (int k = 0; k < DHEAD; k += 32) {
      v16h a = load_a_frag(Qh, DMODEL, qb, hb + k, lane);
      v16h b = load_b_frag(Kh, DMODEL, nt * 16, hb + k, lane);
      acc = __builtin_amdgcn_wmma_f32_16x16x32_f16(
          false, a, false, b, (short)0, acc, false, false);
    }
    int n = lane & 15, hi = lane >> 4;
#pragma unroll
    for (int r = 0; r < 8; ++r)
      sc[(r + 8 * hi) * LDSROW + nt * 16 + n] = acc[r] * 0.125f;
  }
  __syncthreads();

  // ---- phase 2: per-row top-k threshold (bitwise search) + masks + softmax
  for (int rr = 0; rr < 2; ++rr) {
    int row = wave * 2 + rr;
    int qi  = qb + row;
    const float* srow = sc + row * LDSROW;

    // largest T with count(|s| >= T) >= KEEP; |f| compares as its bit pattern
    unsigned lo = 0u, hiB = 0x7F800001u;
    for (int it = 0; it < 32; ++it) {
      unsigned mid = (lo + hiB) >> 1;
      int cnt = 0;
      for (int j = lane; j < SEQ; j += 32) {
        unsigned mb = __float_as_uint(srow[j]) & 0x7FFFFFFFu;
        cnt += (mb >= mid) ? 1 : 0;
      }
      for (int o = 16; o; o >>= 1) cnt += __shfl_xor(cnt, o, 32);
      if (cnt >= KEEP) lo = mid; else hiB = mid;
    }
    unsigned T = lo;

    float mx = -__builtin_inff();
    for (int j = lane; j < SEQ; j += 32) {
      float v = srow[j];
      unsigned mb = __float_as_uint(v) & 0x7FFFFFFFu;
      if ((mb >= T) && (j <= qi + WINDOW)) mx = fmaxf(mx, v);
    }
    for (int o = 16; o; o >>= 1) mx = fmaxf(mx, __shfl_xor(mx, o, 32));

    float sum = 0.f;
    for (int j = lane; j < SEQ; j += 32) {
      float v = srow[j];
      unsigned mb = __float_as_uint(v) & 0x7FFFFFFFu;
      if ((mb >= T) && (j <= qi + WINDOW)) sum += __expf(v - mx);
    }
    for (int o = 16; o; o >>= 1) sum += __shfl_xor(sum, o, 32);
    float inv = 1.f / sum;

    for (int j = lane; j < SEQ; j += 32) {
      float v = srow[j];
      unsigned mb = __float_as_uint(v) & 0x7FFFFFFFu;
      bool keep = (mb >= T) && (j <= qi + WINDOW);
      pr[row * LDSROW + j] = keep ? (_Float16)(__expf(v - mx) * inv)
                                  : (_Float16)0.f;
    }
  }
  __syncthreads();

  // ---- phase 3: ctx = P V, all 8 waves: 4 tiles x 2 K-halves, LDS reduce --
  {
    int tile = wave & 3, kh = wave >> 2;
    v8f acc = zero8();
    for (int ks = kh * (SEQ / 2); ks < (kh + 1) * (SEQ / 2); ks += 32) {
      v16h a = load_a_frag(pr, LDSROW, 0, ks, lane);
      v16h b = load_b_frag(Vth, SEQ, hb + tile * 16, ks, lane);
      acc = __builtin_amdgcn_wmma_f32_16x16x32_f16(
          false, a, false, b, (short)0, acc, false, false);
    }
    float* red = sc;  // score region is dead now; reuse as reduce scratch
    if (kh == 1) {
#pragma unroll
      for (int r = 0; r < 8; ++r) red[(tile * 32 + lane) * 8 + r] = acc[r];
    }
    __syncthreads();
    if (kh == 0) {
      int n = lane & 15, hi = lane >> 4;
#pragma unroll
      for (int r = 0; r < 8; ++r) {
        float v = acc[r] + red[(tile * 32 + lane) * 8 + r];
        Ctx[(size_t)(qb + r + 8 * hi) * DMODEL + hb + tile * 16 + n] = v;
      }
    }
  }
}

// ---------------------------------------------------------------------------

extern "C" void kernel_launch(void* const* d_in, const int* in_sizes, int n_in,
                              void* d_out, int out_size, void* d_ws, size_t ws_size,
                              hipStream_t stream) {
  (void)in_sizes; (void)n_in; (void)out_size; (void)ws_size;

  const float* X      = (const float*)d_in[0];
  const float* A      = (const float*)d_in[1];
  const float* Bm     = (const float*)d_in[2];
  const float* Cm     = (const float*)d_in[3];
  const float* log_dt = (const float*)d_in[4];
  const float* Wg = (const float*)d_in[5];  const float* bg = (const float*)d_in[6];
  const float* Wq = (const float*)d_in[7];  const float* bq = (const float*)d_in[8];
  const float* Wk = (const float*)d_in[9];  const float* bk = (const float*)d_in[10];
  const float* Wv = (const float*)d_in[11]; const float* bv = (const float*)d_in[12];
  const float* Wd = (const float*)d_in[13]; const float* bd = (const float*)d_in[14];

  const size_t SD = (size_t)SEQ * DMODEL;     // 1,572,864
  const size_t DD = (size_t)DMODEL * DMODEL;  //   589,824

  size_t off = 0;
  auto alloc = [&](size_t bytes) -> void* {
    off = (off + 255) & ~(size_t)255;
    void* p = (char*)d_ws + off;
    off += bytes;
    return p;
  };
  _Float16* Xh   = (_Float16*)alloc(SD * 2);
  _Float16* Wgh  = (_Float16*)alloc(DD * 2);
  _Float16* Wqh  = (_Float16*)alloc(DD * 2);
  _Float16* Wkh  = (_Float16*)alloc(DD * 2);
  _Float16* Wvh  = (_Float16*)alloc(DD * 2);
  _Float16* Wdh  = (_Float16*)alloc(DD * 2);
  float*    G    = (float*)   alloc(SD * 4);
  _Float16* Hh   = (_Float16*)alloc(SD * 2);
  _Float16* Qh   = (_Float16*)alloc(SD * 2);
  _Float16* Kh   = (_Float16*)alloc(SD * 2);
  float*    Vf   = (float*)   alloc(SD * 4);
  _Float16* Vth  = (_Float16*)alloc(SD * 2);
  float*    Ctx  = (float*)   alloc(SD * 4);
  _Float16* Ctxh = (_Float16*)alloc(SD * 2);

  const int cvtSD = (int)((SD + 255) / 256);
  const int cvtDD = (int)((DD + 255) / 256);
  cvt_f32_f16<<<cvtSD, 256, 0, stream>>>(X,  Xh,  (int)SD);
  cvt_f32_f16<<<cvtDD, 256, 0, stream>>>(Wg, Wgh, (int)DD);
  cvt_f32_f16<<<cvtDD, 256, 0, stream>>>(Wq, Wqh, (int)DD);
  cvt_f32_f16<<<cvtDD, 256, 0, stream>>>(Wk, Wkh, (int)DD);
  cvt_f32_f16<<<cvtDD, 256, 0, stream>>>(Wv, Wvh, (int)DD);
  cvt_f32_f16<<<cvtDD, 256, 0, stream>>>(Wd, Wdh, (int)DD);

  // one block per 16-row tile, 12 waves = 12 N-strips of 64
  const int gemmBlocks  = SEQ / 16;           // 128
  const int gemmThreads = 32 * (DMODEL / 64); // 384

  // gates: G = sigmoid(X Wg^T + bg)
  gemm_wmma<<<gemmBlocks, gemmThreads, 0, stream>>>(Xh, Wgh, bg, G, nullptr,
                                                    SEQ, DMODEL, DMODEL, 1);
  // SSM scan -> Hh (f16)
  scan_kernel<<<DMODEL / 16, 256, 0, stream>>>(X, A, Bm, Cm, log_dt, G, Hh);

  // projections
  gemm_wmma<<<gemmBlocks, gemmThreads, 0, stream>>>(Hh, Wqh, bq, nullptr, Qh,
                                                    SEQ, DMODEL, DMODEL, 0);
  gemm_wmma<<<gemmBlocks, gemmThreads, 0, stream>>>(Hh, Wkh, bk, nullptr, Kh,
                                                    SEQ, DMODEL, DMODEL, 0);
  gemm_wmma<<<gemmBlocks, gemmThreads, 0, stream>>>(Hh, Wvh, bv, Vf, nullptr,
                                                    SEQ, DMODEL, DMODEL, 0);
  transpose_cvt<<<cvtSD, 256, 0, stream>>>(Vf, Vth, SEQ, DMODEL);

  // attention: grid (query blocks, heads); 198144 B dynamic LDS (<320KB/WGP)
  dim3 agrid(SEQ / 16, NHEAD);
  size_t shmem = (size_t)16 * LDSROW * sizeof(float)
               + (size_t)16 * LDSROW * sizeof(_Float16);
  attn_kernel<<<agrid, 256, shmem, stream>>>(Qh, Kh, Vth, Ctx);

  // output projection -> d_out (fp32)
  cvt_f32_f16<<<cvtSD, 256, 0, stream>>>(Ctx, Ctxh, (int)SD);
  gemm_wmma<<<gemmBlocks, gemmThreads, 0, stream>>>(Ctxh, Wdh, bd,
                                                    (float*)d_out, nullptr,
                                                    SEQ, DMODEL, DMODEL, 0);
}
